// MutualInformation_75849122447882
// MI455X (gfx1250) — matrix operand order repaired and checked by reading, not compiled, not run on hardware
//
#include <hip/hip_runtime.h>
#include <hip/hip_bf16.h>

#define NUM_BINS 64
#define CHUNK 128
#define PAD 8
#define ROWSTRIDE (CHUNK + PAD)
#define MI_EPS 1e-8f

typedef _Float16 v16h __attribute__((ext_vector_type(16)));
typedef _Float16 v8h  __attribute__((ext_vector_type(8)));
typedef float    v8f  __attribute__((ext_vector_type(8)));

// Gaussian recurrence constants for bins k/63, sigma=0.1 (u = 1/sigma^2 = 100):
//   w_0 = exp(-50*x^2), r_0 = exp(u*h*x - 0.5*u*h^2), q = exp(-u*h^2)
#define GH_UH    1.5873016f     // u*h = 100/63
#define GH_HALF  0.012597631f   // 0.5*u*h^2
#define GH_Q     0.97511947f    // exp(-u*h^2)

// ws layout (floats):
//   [0,     8192)  joint accumulators, B=2 x 64 x 64
//   [8192,  8320)  p_f sums, 2 x 64
//   [8320,  8448)  p_w sums, 2 x 64
//   [8448,  8456)  min/max as uint bits: fmin[2], fmax[2], wmin[2], wmax[2]
#define WS_JOINT 0
#define WS_PF    8192
#define WS_PW    8320
#define WS_MM    8448

__global__ void mi_init(float* ws) {
    int t = blockIdx.x * blockDim.x + threadIdx.x;
    for (int i = t; i < WS_MM; i += gridDim.x * blockDim.x) ws[i] = 0.0f;
    if (blockIdx.x == 0 && threadIdx.x < 8) {
        unsigned* mm = (unsigned*)(ws + WS_MM);
        bool isMin = (threadIdx.x < 2) || (threadIdx.x >= 4 && threadIdx.x < 6);
        mm[threadIdx.x] = isMin ? 0x7F7FFFFFu : 0u;   // FLT_MAX bits / 0.0f bits
    }
}

// Inputs are uniform[0,1): non-negative floats => bit pattern is order-monotonic,
// so integer atomicMin/Max on the bits gives exact float min/max.
__global__ void mi_minmax(const float* __restrict__ fixedp,
                          const float* __restrict__ warpedp,
                          float* ws, int Nelem) {
    unsigned* mm = (unsigned*)(ws + WS_MM);
    float fmn[2] = {3.4e38f, 3.4e38f}, fmx[2] = {0.0f, 0.0f};
    float wmn[2] = {3.4e38f, 3.4e38f}, wmx[2] = {0.0f, 0.0f};
    long total = 2L * (long)Nelem;
    for (long i = (long)blockIdx.x * blockDim.x + threadIdx.x; i < total;
         i += (long)gridDim.x * blockDim.x) {
        int b = (int)(i / Nelem);
        float f = fixedp[i], w = warpedp[i];
        fmn[b] = fminf(fmn[b], f); fmx[b] = fmaxf(fmx[b], f);
        wmn[b] = fminf(wmn[b], w); wmx[b] = fmaxf(wmx[b], w);
    }
    for (int off = 16; off > 0; off >>= 1) {
        #pragma unroll
        for (int b = 0; b < 2; b++) {
            fmn[b] = fminf(fmn[b], __shfl_xor(fmn[b], off));
            fmx[b] = fmaxf(fmx[b], __shfl_xor(fmx[b], off));
            wmn[b] = fminf(wmn[b], __shfl_xor(wmn[b], off));
            wmx[b] = fmaxf(wmx[b], __shfl_xor(wmx[b], off));
        }
    }
    if ((threadIdx.x & 31) == 0) {
        #pragma unroll
        for (int b = 0; b < 2; b++) {
            atomicMin(&mm[0 + b], __float_as_uint(fmn[b]));
            atomicMax(&mm[2 + b], __float_as_uint(fmx[b]));
            atomicMin(&mm[4 + b], __float_as_uint(wmn[b]));
            atomicMax(&mm[6 + b], __float_as_uint(wmx[b]));
        }
    }
}

// Main kernel: per stage of 128 voxels:
//  pass1: each thread generates its voxel's 64 unnormalized Gaussian weights via
//         a 2-mul/bin recurrence (2 exps total), stores f16 transposed [bin][voxel],
//         records 1/sum in LDS.
//  pass2: row-owning threads normalize in packed f16 (v_pk_mul_f16) and fuse the
//         marginal (p_f/p_w) accumulation.
//  WMMA:  8 waves x 2 tiles x 4 K-steps of v_wmma_f32_16x16x32_f16.
__global__ void __launch_bounds__(256)
mi_main(const float* __restrict__ fixedp, const float* __restrict__ warpedp,
        float* ws, int Nelem) {
    __shared__ _Float16 sWf[NUM_BINS][ROWSTRIDE];  // fixed weights, [bin][voxel]
    __shared__ _Float16 sWw[NUM_BINS][ROWSTRIDE];  // warped weights, [bin][voxel]
    __shared__ _Float16 sInv[2][CHUNK];            // per-voxel 1/(sum+eps), f16

    const int b    = blockIdx.y;
    const int t    = threadIdx.x;
    const int wave = t >> 5;
    const int lane = t & 31;
    const int tk   = wave & 3;          // tile row (bins of fixed)
    const int tl0  = (wave >> 2) * 2;   // tile cols (bins of warped): tl0, tl0+1

    const unsigned* mm = (const unsigned*)(ws + WS_MM);
    const float fmin = __uint_as_float(mm[0 + b]);
    const float fmax = __uint_as_float(mm[2 + b]);
    const float wmin = __uint_as_float(mm[4 + b]);
    const float wmax = __uint_as_float(mm[6 + b]);
    const float fScale = 1.0f / (fmax - fmin + MI_EPS);
    const float wScale = 1.0f / (wmax - wmin + MI_EPS);

    // pass1 role: t<128 -> fixed voxel t; t>=128 -> warped voxel t-128
    const bool  isF  = (t < CHUNK);
    const float mn   = isF ? fmin : wmin;
    const float sc   = isF ? fScale : wScale;
    const float* img = isF ? fixedp : warpedp;
    _Float16* srow   = isF ? &sWf[0][0] : &sWw[0][0];
    _Float16* sinvp  = isF ? &sInv[0][0] : &sInv[1][0];
    const int cidx   = t & (CHUNK - 1);

    // pass2 role: thread owns half of one [image,bin] row
    const int rowId  = t & 127;            // 0..63 fixed bins, 64..127 warped bins
    const int half   = t >> 7;             // which half of the row (cols 0..63 / 64..127)
    _Float16* prow         = (rowId < NUM_BINS) ? &sWf[rowId][0] : &sWw[rowId - NUM_BINS][0];
    const _Float16* prinv  = (rowId < NUM_BINS) ? &sInv[0][0]    : &sInv[1][0];

    v8f acc0 = {0,0,0,0,0,0,0,0};
    v8f acc1 = {0,0,0,0,0,0,0,0};
    float margLoc = 0.0f;

    // fragment index precompute (A 16x32 & B 32x16 fragments, contiguous k = 16*(lane/16) + s in both)
    const int mrow = tk * 16 + (lane & 15);
    const int aoff = (lane < 16) ? 0 : 16;
    const int lA   = tl0 * 16 + (lane & 15);
    const int lB   = lA + 16;
    const int boff = (lane < 16) ? 0 : 16;

    for (long base = (long)blockIdx.x * CHUNK; base < Nelem;
         base += (long)gridDim.x * CHUNK) {
        // ---- pass1: unnormalized Gaussian weights via multiplicative recurrence ----
        long e = base + cidx;
        bool valid = (e < Nelem);
        float x  = valid ? img[(long)b * Nelem + e] : 0.0f;
        float xn = (x - mn) * sc;
        float w = valid ? __expf(-50.0f * xn * xn) : 0.0f;   // bin 0
        float r = __expf(GH_UH * xn - GH_HALF);              // ratio to next bin
        float s = 0.0f;
        #pragma unroll
        for (int k = 0; k < NUM_BINS; k++) {
            s += w;
            srow[k * ROWSTRIDE + cidx] = (_Float16)w;
            w *= r;
            r *= GH_Q;
        }
        sinvp[cidx] = valid ? (_Float16)(1.0f / (s + MI_EPS)) : (_Float16)0.0f;
        __syncthreads();

        // ---- pass2: normalize rows in packed f16, fuse marginal accumulation ----
        {
            float msum = 0.0f;
            #pragma unroll
            for (int j = 0; j < 8; j++) {
                const int c0 = (half * 8 + j) * 8;
                v8h v  = *(const v8h*)&prow[c0];
                v8h iv = *(const v8h*)&prinv[c0];
                v = v * iv;                       // v_pk_mul_f16
                *(v8h*)&prow[c0] = v;
                #pragma unroll
                for (int q = 0; q < 8; q++) msum += (float)v[q];
            }
            margLoc += msum;
        }
        __syncthreads();

        // ---- joint histogram: 4 K-chunks of 32 voxels via WMMA ----
        #pragma unroll
        for (int kc = 0; kc < CHUNK / 32; kc++) {
            const int n0 = kc * 32;
            union { v16h v; v8h h[2]; } a, b0, b1;
            a.h[0]  = *(const v8h*)&sWf[mrow][n0 + aoff];
            a.h[1]  = *(const v8h*)&sWf[mrow][n0 + aoff + 8];
            b0.h[0] = *(const v8h*)&sWw[lA][n0 + boff];
            b0.h[1] = *(const v8h*)&sWw[lA][n0 + boff + 8];
            b1.h[0] = *(const v8h*)&sWw[lB][n0 + boff];
            b1.h[1] = *(const v8h*)&sWw[lB][n0 + boff + 8];
            acc0 = __builtin_amdgcn_wmma_f32_16x16x32_f16(
                       false, a.v, false, b0.v, (short)0, acc0, false, false);
            acc1 = __builtin_amdgcn_wmma_f32_16x16x32_f16(
                       false, a.v, false, b1.v, (short)0, acc1, false, false);
        }
        __syncthreads();
    }

    // ---- flush marginals (both half-row owners atomically combine) ----
    if (rowId < NUM_BINS) atomicAdd(ws + WS_PF + b * 64 + rowId, margLoc);
    else                  atomicAdd(ws + WS_PW + b * 64 + (rowId - NUM_BINS), margLoc);

    // ---- flush joint tiles ----
    float* joint = ws + WS_JOINT + b * 4096;
    const int Nc    = lane & 15;
    const int mBase = (lane < 16) ? 0 : 8;
    #pragma unroll
    for (int rr = 0; rr < 8; rr++) {
        int kRow = tk * 16 + mBase + rr;
        atomicAdd(&joint[kRow * 64 + tl0 * 16 + Nc],       acc0[rr]);
        atomicAdd(&joint[kRow * 64 + (tl0 + 1) * 16 + Nc], acc1[rr]);
    }
}

__global__ void mi_finalize(const float* ws, float* out, float invN) {
    __shared__ float red[256];
    __shared__ float miAcc;
    const int t = threadIdx.x;
    if (t == 0) miAcc = 0.0f;
    __syncthreads();
    for (int b = 0; b < 2; b++) {
        const float* joint = ws + WS_JOINT + b * 4096;
        const float* pf    = ws + WS_PF + b * 64;
        const float* pw    = ws + WS_PW + b * 64;
        float s = 0.0f;
        for (int i = t; i < 4096; i += 256) s += joint[i];
        red[t] = s; __syncthreads();
        for (int o = 128; o > 0; o >>= 1) { if (t < o) red[t] += red[t + o]; __syncthreads(); }
        float denom = red[0] * invN + MI_EPS;
        __syncthreads();
        float mi = 0.0f;
        for (int i = t; i < 4096; i += 256) {
            float pj = (joint[i] * invN) / denom;
            int r = i >> 6, c = i & 63;
            float pp = (pf[r] * invN) * (pw[c] * invN);
            mi += pj * __logf((pj + MI_EPS) / (pp + MI_EPS));
        }
        red[t] = mi; __syncthreads();
        for (int o = 128; o > 0; o >>= 1) { if (t < o) red[t] += red[t + o]; __syncthreads(); }
        if (t == 0) miAcc += red[0];
        __syncthreads();
    }
    if (t == 0) out[0] = -0.5f * miAcc;   // mean over B=2, negated
}

extern "C" void kernel_launch(void* const* d_in, const int* in_sizes, int n_in,
                              void* d_out, int out_size, void* d_ws, size_t ws_size,
                              hipStream_t stream) {
    const float* fixedp  = (const float*)d_in[0];
    const float* warpedp = (const float*)d_in[1];
    float* ws  = (float*)d_ws;
    float* out = (float*)d_out;
    const int B = 2;
    const int Nelem = in_sizes[0] / B;   // 96^3 = 884736 per batch

    mi_init<<<32, 256, 0, stream>>>(ws);
    mi_minmax<<<512, 256, 0, stream>>>(fixedp, warpedp, ws, Nelem);
    dim3 grid(432, B);                    // grid-stride stages of 128 voxels
    mi_main<<<grid, 256, 0, stream>>>(fixedp, warpedp, ws, Nelem);
    mi_finalize<<<1, 256, 0, stream>>>(ws, out, 1.0f / (float)Nelem);
}
